// LinearFLH_59811714564396
// MI455X (gfx1250) — matrix operand order repaired and checked by reading, not compiled, not run on hardware
//
#include <hip/hip_runtime.h>

typedef _Float16 f16x2 __attribute__((ext_vector_type(2)));
typedef _Float16 f16x8 __attribute__((ext_vector_type(8)));
typedef _Float16 v16h  __attribute__((ext_vector_type(16)));
typedef float    v8f   __attribute__((ext_vector_type(8)));
typedef float    f32x4 __attribute__((ext_vector_type(4)));
typedef int      i32x4 __attribute__((ext_vector_type(4)));

#define M_TOTAL 1024
#define N_TOTAL 4096
#define K_TOTAL 4096
#define BM 128
#define BN 128
#define BK 64
#define NGROUPS 32
#define LDSK 72      // f16 per LDS row: 64 data + 8 pad -> 144B stride (16B aligned, 16-row bank spread)
#define NT (K_TOTAL / BK)

// flat (generic) pointer to a __shared__ object: low 32 bits are the LDS offset
static __device__ __forceinline__ unsigned lds_off(const void* p) {
    return (unsigned)(unsigned long long)p;
}

// CDNA5 async copy global -> LDS (ASYNCcnt); VDST vgpr carries per-lane LDS byte address
static __device__ __forceinline__ void async_copy_b128(unsigned lds, const _Float16* g) {
    asm volatile("global_load_async_to_lds_b128 %0, %1, off"
                 :: "v"(lds), "v"(g) : "memory");
}
static __device__ __forceinline__ void wait_async0() {
    asm volatile("s_wait_asynccnt 0x0" ::: "memory");
}

// ---- pre-pass: x f32 -> f16 with per-8-group k-permutation [0,4,1,5,2,6,3,7] ----
__global__ __launch_bounds__(256)
void x_to_f16_perm(const float* __restrict__ x, _Float16* __restrict__ xh) {
    size_t i = ((size_t)blockIdx.x * 256 + threadIdx.x) * 8;
    f32x4 a = *(const f32x4*)(x + i);
    f32x4 b = *(const f32x4*)(x + i + 4);
    f16x8 h;
#pragma unroll
    for (int j = 0; j < 4; ++j) {
        h[2 * j]     = (_Float16)a[j];   // true k = j
        h[2 * j + 1] = (_Float16)b[j];   // true k = j + 4
    }
    *(f16x8*)(xh + i) = h;
}

// dequant one packed word -> 8 f16 in permuted order (pairs (j, j+4)) via f16 magic numbers
static __device__ __forceinline__ f16x8 dq_word(unsigned w, f16x2 c2, f16x2 s2) {
    unsigned t = w ^ 0x88888888u;        // v^8: signed nibble sv = (v^8) - 8
    f16x8 r;
#pragma unroll
    for (int j = 0; j < 4; ++j) {
        unsigned u = ((t >> (4 * j)) & 0x000F000Fu) | 0x64006400u; // packed f16 (1024+(v^8))
        f16x2 uh = __builtin_bit_cast(f16x2, u);
        f16x2 d  = (uh - c2) * s2;       // ((1024+v^8) - (1032+zp)) * sc = (sv - zp)*sc
        r[2 * j]     = d[0];
        r[2 * j + 1] = d[1];
    }
    return r;
}

__global__ __launch_bounds__(256)
void w4a16_wmma_gemm(const _Float16* __restrict__ xh,
                     const unsigned* __restrict__ wq,
                     const float*    __restrict__ wscale,
                     const int*      __restrict__ wzero,
                     float*          __restrict__ out)
{
    __shared__ __attribute__((aligned(16))) _Float16 As[2][BM][LDSK]; // 36 KB
    __shared__ __attribute__((aligned(16))) _Float16 Bs[2][BN][LDSK]; // 36 KB

    const int tid  = threadIdx.x;
    const int lane = tid & 31;
    const int wave = tid >> 5;       // 8 waves: 2(M) x 4(N); wave tile 64x32
    const int wm   = wave & 1;
    const int wn   = wave >> 1;
    const int mBase = blockIdx.y * BM;
    const int nBase = blockIdx.x * BN;

    // staging: thread covers row tid/2, k-half (tid&1)*32 (32 f16 of x / 4 packed words of W)
    const int rowS = tid >> 1;
    const int kh32 = (tid & 1) * 32;

    const _Float16* xRow = xh + (size_t)(mBase + rowS) * K_TOTAL + kh32;
    const unsigned* wRow = wq + (size_t)(nBase + rowS) * (K_TOTAL / 8) + (tid & 1) * 4;
    const float*    sRow = wscale + (size_t)(nBase + rowS) * NGROUPS;
    const int*      zRow = wzero  + (size_t)(nBase + rowS) * NGROUPS;

    auto issue_async_A = [&](int kt, int buf) {
#pragma unroll
        for (int i = 0; i < 4; ++i)
            async_copy_b128(lds_off(&As[buf][rowS][kh32 + i * 8]),
                            xRow + (size_t)kt * BK + i * 8);
    };

    i32x4 wv;
    float sc;
    int   zp;
    auto load_B = [&](int kt) {
        wv = *(const i32x4*)(wRow + kt * (BK / 8));
        const int g = kt >> 1;           // 128-elem group = 2 K-steps
        sc = sRow[g];
        zp = zRow[g];
    };

    auto store_Bs = [&](int buf) {
        const _Float16 ch = (_Float16)(float)(1032 + zp);
        const _Float16 sh = (_Float16)sc;
        const f16x2 c2 = {ch, ch};
        const f16x2 s2 = {sh, sh};
#pragma unroll
        for (int w = 0; w < 4; ++w)
            *(f16x8*)&Bs[buf][rowS][kh32 + w * 8] = dq_word((unsigned)wv[w], c2, s2);
    };

    v8f acc[4][2] = {};

    auto compute = [&](int buf) {
        const int ar = lane & 15;
        const int ak = (lane >> 4) << 3;   // A lane: slots ak..ak+7 and ak+16..ak+23
        const int bk = (lane >> 4) << 3;   // B lane: slots bk..bk+7 and bk+16..bk+23 (column = ar)
#pragma unroll
        for (int h = 0; h < 2; ++h) {      // two K=32 halves of the K=64 step
            v16h a[4], b[2];
#pragma unroll
            for (int fa = 0; fa < 4; ++fa) {
                const _Float16* ap = &As[buf][wm * 64 + fa * 16 + ar][h * 32 + ak];
                f16x8 lo = *(const f16x8*)ap;
                f16x8 hi = *(const f16x8*)(ap + 16);
                a[fa] = __builtin_shufflevector(lo, hi, 0,1,2,3,4,5,6,7,8,9,10,11,12,13,14,15);
            }
#pragma unroll
            for (int fb = 0; fb < 2; ++fb) {
                const _Float16* bp = &Bs[buf][wn * 32 + fb * 16 + ar][h * 32 + bk];
                f16x8 lo = *(const f16x8*)bp;
                f16x8 hi = *(const f16x8*)(bp + 16);
                b[fb] = __builtin_shufflevector(lo, hi, 0,1,2,3,4,5,6,7,8,9,10,11,12,13,14,15);
            }
#pragma unroll
            for (int fa = 0; fa < 4; ++fa)
#pragma unroll
                for (int fb = 0; fb < 2; ++fb)
                    acc[fa][fb] = __builtin_amdgcn_wmma_f32_16x16x32_f16(
                        false, a[fa], false, b[fb], (short)0, acc[fa][fb], false, false);
        }
    };

    // prologue
    issue_async_A(0, 0);
    load_B(0);
    store_Bs(0);

    for (int kt = 0; kt < NT; ++kt) {
        const int cur = kt & 1;
        wait_async0();                    // our async writes into buf `cur` done
        __syncthreads();                  // everyone's writes visible; prev reads of buf cur^1 done
        if (kt + 1 < NT) {
            load_B(kt + 1);               // global loads overlap WMMA
            issue_async_A(kt + 1, cur ^ 1);
        }
        compute(cur);
        if (kt + 1 < NT) store_Bs(cur ^ 1);
    }

    // writeback: D layout VGPR r <-> M = (lane>>4)*8 + r, N = lane&15
    const int cn  = nBase + wn * 32 + (lane & 15);
    const int cm0 = mBase + wm * 64 + ((lane >> 4) << 3);
#pragma unroll
    for (int fa = 0; fa < 4; ++fa)
#pragma unroll
        for (int fb = 0; fb < 2; ++fb) {
            const int n  = cn + fb * 16;
            const int m0 = cm0 + fa * 16;
#pragma unroll
            for (int r = 0; r < 8; ++r)
                out[(size_t)(m0 + r) * N_TOTAL + n] = acc[fa][fb][r];
        }
}

extern "C" void kernel_launch(void* const* d_in, const int* in_sizes, int n_in,
                              void* d_out, int out_size, void* d_ws, size_t ws_size,
                              hipStream_t stream) {
    const float* x  = (const float*)d_in[0];   // [1024,4096] f32
    const int*   wq = (const int*)d_in[1];     // [2097152] packed int4
    const float* ws = (const float*)d_in[2];   // [4096,32]
    const int*   wz = (const int*)d_in[3];     // [4096,32]
    float* out = (float*)d_out;                // [1024,4096] f32

    _Float16* xh = (_Float16*)d_ws;            // 1024*4096 f16 = 8.4 MB scratch

    const int nconv = (M_TOTAL * K_TOTAL / 8) / 256;  // 2048 blocks
    x_to_f16_perm<<<nconv, 256, 0, stream>>>(x, xh);

    dim3 grid(N_TOTAL / BN, M_TOTAL / BM);     // (32, 8)
    w4a16_wmma_gemm<<<grid, 256, 0, stream>>>(xh, (const unsigned*)wq, ws, wz, out);
}
